// GatedScan_42606075576714
// MI455X (gfx1250) — matrix operand order repaired and checked by reading, not compiled, not run on hardware
//
#include <hip/hip_runtime.h>
#include <hip/hip_bf16.h>

// ---------------------------------------------------------------------------
// GatedScan for MI455X (gfx1250, wave32, WMMA).
// B=4, S=4096, D_IN=1024, D_ST=1024.  M = B*S = 16384.
// GEMMs in bf16 via v_wmma_f32_16x16x32_bf16 (4 M-tiles per wave, B reuse).
// A and B operands pre-packed into exact WMMA fragment lane layouts.
// Scan in fp32, 3-pass chunked; apply pass emits h directly in packed-A bf16.
// ---------------------------------------------------------------------------

#define BB    4
#define SS    4096
#define DIN   1024
#define DST   1024
#define MM    (BB * SS)          // 16384 rows
#define CH    64                 // scan chunks
#define CL    64                 // chunk length (CH*CL == SS)
#define MBLK  4                  // M-tiles (of 16 rows) per wave in GEMMs

typedef __attribute__((ext_vector_type(16))) __bf16 v16bf;
typedef __attribute__((ext_vector_type(8)))  float  v8f;

__device__ __forceinline__ __bf16 f2bf(float f) {
    unsigned int u = __float_as_uint(f);
    u += 0x7FFFu + ((u >> 16) & 1u);            // round-to-nearest-even
    unsigned short s = (unsigned short)(u >> 16);
    __bf16 r;
    __builtin_memcpy(&r, &s, 2);
    return r;
}

// -------------------------- fragment packing -------------------------------
// A-fragment layout (16-bit A 16x32, wave32):
//   packed idx = ((mt*(K/32)+kc)*32 + lane)*16 + i
//   m = mt*16 + (lane&15)
//   k = kc*32 + (lane>>4)*8 + i + (i>=8 ? 8 : 0)
// B-fragment layout (16-bit B 32x16, wave32):
//   packed idx = ((nt*(K/32)+kc)*32 + lane)*16 + i
//   n = nt*16 + (lane&15),  k = kc*32 + (lane>>4)*8 + i + (i>=8 ? 8 : 0)

__global__ void pack_a(const float* __restrict__ src,   // row-major [M x K] fp32
                       __bf16* __restrict__ dst, int K) {
    int idx  = blockIdx.x * blockDim.x + threadIdx.x;
    int i    = idx & 15;
    int lane = (idx >> 4) & 31;
    int kc   = (idx >> 9) & ((K / 32) - 1);
    int mt   = idx / (512 * (K / 32));
    int m = mt * 16 + (lane & 15);
    int k = kc * 32 + (lane >> 4) * 8 + i + ((i >= 8) ? 8 : 0);
    dst[idx] = f2bf(src[(size_t)m * K + k]);
}

__global__ void pack_w(const float* __restrict__ w,     // row-major [K x N] fp32
                       __bf16* __restrict__ wp, int K, int N) {
    int idx  = blockIdx.x * blockDim.x + threadIdx.x;
    int i    = idx & 15;
    int lane = (idx >> 4) & 31;
    int kc   = (idx >> 9) & ((K / 32) - 1);
    int nt   = idx / (512 * (K / 32));
    int n = nt * 16 + (lane & 15);
    int k = kc * 32 + (lane >> 4) * 8 + i + ((i >= 8) ? 8 : 0);
    wp[idx] = f2bf(w[(size_t)k * N + n]);
}

// -------------------------- fused gate/inp GEMM ----------------------------
// Each wave computes a 64x16 tile of BOTH gate and inp (B fragments reused 4x).
__global__ void gemm_gate_inp(const __bf16* __restrict__ xap,   // packed A
                              const __bf16* __restrict__ wg,    // packed B
                              const __bf16* __restrict__ wi,    // packed B
                              const float*  __restrict__ gate_b,
                              const float*  __restrict__ inp_b,
                              float* __restrict__ aOut,
                              float* __restrict__ bOut) {
    constexpr int K  = DIN;
    constexpr int N  = DST;
    constexpr int KC = K / 32;
    constexpr int NT = N / 16;
    const int lane    = threadIdx.x & 31;
    const int wave    = threadIdx.x >> 5;
    const int tile    = blockIdx.x * (blockDim.x >> 5) + wave;
    const int mb      = tile / NT;          // 64-row block index
    const int nt      = tile % NT;
    const int halfSel = lane >> 4;
    const int l15     = lane & 15;

    v8f accG[MBLK] = {};
    v8f accI[MBLK] = {};
    for (int kc = 0; kc < KC; ++kc) {
        const v16bf bg = *(const v16bf*)(wg + ((size_t)(nt * KC + kc) * 32 + lane) * 16);
        const v16bf bi = *(const v16bf*)(wi + ((size_t)(nt * KC + kc) * 32 + lane) * 16);
#pragma unroll
        for (int t = 0; t < MBLK; ++t) {
            const int mt = mb * MBLK + t;
            const v16bf av = *(const v16bf*)(xap + ((size_t)(mt * KC + kc) * 32 + lane) * 16);
            accG[t] = __builtin_amdgcn_wmma_f32_16x16x32_bf16(false, av, false, bg,
                                                              (short)0, accG[t], false, false);
            accI[t] = __builtin_amdgcn_wmma_f32_16x16x32_bf16(false, av, false, bi,
                                                              (short)0, accI[t], false, false);
        }
    }

    const int   n  = nt * 16 + l15;
    const float gb = gate_b[n];
    const float ib = inp_b[n];
#pragma unroll
    for (int t = 0; t < MBLK; ++t) {
#pragma unroll
        for (int r = 0; r < 8; ++r) {
            const int   m = (mb * MBLK + t) * 16 + halfSel * 8 + r;
            const float g = 1.f / (1.f + __expf(-(accG[t][r] + gb)));
            const float v = (1.f - g) * (accI[t][r] + ib);
            const size_t o = (size_t)m * N + n;
            aOut[o] = g;
            bOut[o] = v;
        }
    }
}

// -------------------------- 3-pass chunked scan ----------------------------
// h_t = a_t * h_{t-1} + b_t along S, independent per (batch, channel).

__global__ void scan_chunk(const float* __restrict__ a,
                           const float* __restrict__ b,
                           float* __restrict__ aggA,
                           float* __restrict__ aggB) {
    int tid   = blockIdx.x * blockDim.x + threadIdx.x;   // BB*CH*DST threads
    int d     = tid & (DST - 1);
    int t2    = tid >> 10;
    int chunk = t2 & (CH - 1);
    int bb    = t2 >> 6;
    size_t base = ((size_t)bb * SS + (size_t)chunk * CL) * DST + d;
    float aP = 1.f, bP = 0.f;
    for (int t = 0; t < CL; ++t) {
        float A  = a[base + (size_t)t * DST];
        float Bv = b[base + (size_t)t * DST];
        bP = A * bP + Bv;
        aP = A * aP;
    }
    size_t o = ((size_t)bb * CH + chunk) * DST + d;
    aggA[o] = aP;
    aggB[o] = bP;
}

__global__ void scan_carry(const float* __restrict__ aggA,
                           const float* __restrict__ aggB,
                           float* __restrict__ carry) {
    int tid = blockIdx.x * blockDim.x + threadIdx.x;     // BB*DST threads
    int d   = tid & (DST - 1);
    int bb  = tid >> 10;
    float h = 0.f;
    for (int c = 0; c < CH; ++c) {
        size_t o = ((size_t)bb * CH + c) * DST + d;
        carry[o] = h;
        h = aggA[o] * h + aggB[o];
    }
}

// Applies carries and writes h directly in packed-A bf16 layout (K = DST).
__global__ void scan_apply(const float* __restrict__ a,
                           const float* __restrict__ b,
                           const float* __restrict__ carry,
                           __bf16* __restrict__ hp) {
    constexpr int KC = DST / 32;
    int tid   = blockIdx.x * blockDim.x + threadIdx.x;   // BB*CH*DST threads
    int d     = tid & (DST - 1);
    int t2    = tid >> 10;
    int chunk = t2 & (CH - 1);
    int bb    = t2 >> 6;
    size_t base = ((size_t)bb * SS + (size_t)chunk * CL) * DST + d;

    // packed-A coordinates for this channel d (fixed per thread)
    const int kc  = d >> 5;
    const int kk  = d & 31;
    const int hs  = (kk >> 3) & 1;                        // lane half
    const int ip  = (kk & 7) + ((kk >= 16) ? 8 : 0);      // element index
    const int lb  = hs * 16;                              // lane base

    float h = carry[((size_t)bb * CH + chunk) * DST + d];
    const int m0 = bb * SS + chunk * CL;
    for (int t = 0; t < CL; ++t) {
        size_t o = base + (size_t)t * DST;
        h = a[o] * h + b[o];
        const int m    = m0 + t;
        const int mt   = m >> 4;
        const int mrow = m & 15;
        hp[((size_t)(mt * KC + kc) * 32 + (lb + mrow)) * 16 + ip] = f2bf(h);
    }
}

// -------------------------- output GEMM + residual -------------------------

__global__ void gemm_out(const __bf16* __restrict__ hp,   // packed A
                         const __bf16* __restrict__ wo,   // packed B
                         const float*  __restrict__ x,
                         const float*  __restrict__ out_b,
                         float* __restrict__ out) {
    constexpr int K  = DST;
    constexpr int N  = DIN;
    constexpr int KC = K / 32;
    constexpr int NT = N / 16;
    const int lane    = threadIdx.x & 31;
    const int wave    = threadIdx.x >> 5;
    const int tile    = blockIdx.x * (blockDim.x >> 5) + wave;
    const int mb      = tile / NT;
    const int nt      = tile % NT;
    const int halfSel = lane >> 4;
    const int l15     = lane & 15;

    v8f acc[MBLK] = {};
    for (int kc = 0; kc < KC; ++kc) {
        const v16bf bv = *(const v16bf*)(wo + ((size_t)(nt * KC + kc) * 32 + lane) * 16);
#pragma unroll
        for (int t = 0; t < MBLK; ++t) {
            const int mt = mb * MBLK + t;
            const v16bf av = *(const v16bf*)(hp + ((size_t)(mt * KC + kc) * 32 + lane) * 16);
            acc[t] = __builtin_amdgcn_wmma_f32_16x16x32_bf16(false, av, false, bv,
                                                             (short)0, acc[t], false, false);
        }
    }

    const int   n  = nt * 16 + l15;
    const float ob = out_b[n];
#pragma unroll
    for (int t = 0; t < MBLK; ++t) {
#pragma unroll
        for (int r = 0; r < 8; ++r) {
            const int   m = (mb * MBLK + t) * 16 + halfSel * 8 + r;
            const size_t o = (size_t)m * N + n;
            out[o] = x[o] + acc[t][r] + ob;
        }
    }
}

// -------------------------- launcher ---------------------------------------

extern "C" void kernel_launch(void* const* d_in, const int* in_sizes, int n_in,
                              void* d_out, int out_size, void* d_ws, size_t ws_size,
                              hipStream_t stream) {
    const float* x  = (const float*)d_in[0];
    const float* gw = (const float*)d_in[1];
    const float* gb = (const float*)d_in[2];
    const float* iw = (const float*)d_in[3];
    const float* ib = (const float*)d_in[4];
    const float* ow = (const float*)d_in[5];
    const float* ob = (const float*)d_in[6];
    float* out = (float*)d_out;

    char*  ws  = (char*)d_ws;
    size_t off = 0;
    auto alloc = [&](size_t bytes) -> void* {
        off = (off + 255) & ~(size_t)255;
        void* p = ws + off;
        off += bytes;
        return p;
    };

    __bf16* xap  = (__bf16*)alloc((size_t)MM * DIN * 2);   // 32 MB packed A (x)
    __bf16* wgp  = (__bf16*)alloc((size_t)DIN * DST * 2);  //  2 MB packed B
    __bf16* wip  = (__bf16*)alloc((size_t)DIN * DST * 2);  //  2 MB packed B
    __bf16* wop  = (__bf16*)alloc((size_t)DST * DIN * 2);  //  2 MB packed B
    float*  aArr = (float*)alloc((size_t)MM * DST * 4);    // 64 MB gate
    float*  bArr = (float*)alloc((size_t)MM * DST * 4);    // 64 MB (1-g)*inp
    __bf16* hp   = (__bf16*)alloc((size_t)MM * DST * 2);   // 32 MB packed A (h)
    float*  aggA = (float*)alloc((size_t)BB * CH * DST * 4);
    float*  aggB = (float*)alloc((size_t)BB * CH * DST * 4);
    float*  cry  = (float*)alloc((size_t)BB * CH * DST * 4);
    (void)ws_size; (void)in_sizes; (void)n_in; (void)out_size;

    // 1) pack operands (fp32 -> bf16 fragments)
    {
        int na = MM * DIN;
        pack_a<<<na / 256, 256, 0, stream>>>(x, xap, DIN);
        int wn = DIN * DST;
        pack_w<<<(wn + 255) / 256, 256, 0, stream>>>(gw, wgp, DIN, DST);
        pack_w<<<(wn + 255) / 256, 256, 0, stream>>>(iw, wip, DIN, DST);
        pack_w<<<(wn + 255) / 256, 256, 0, stream>>>(ow, wop, DST, DIN);
    }

    // 2) fused gate/inp GEMM (WMMA), sigmoid/blend epilogue -> aArr, bArr
    {
        int tiles = (MM / (16 * MBLK)) * (DST / 16);   // 16384 wave-tiles
        gemm_gate_inp<<<tiles / 4, 128, 0, stream>>>(xap, wgp, wip, gb, ib, aArr, bArr);
    }

    // 3) chunked affine scan -> hp (packed-A bf16)
    {
        int n1 = BB * CH * DST;                        // 262144
        scan_chunk<<<n1 / 256, 256, 0, stream>>>(aArr, bArr, aggA, aggB);
        int n2 = BB * DST;                             // 4096
        scan_carry<<<n2 / 256, 256, 0, stream>>>(aggA, aggB, cry);
        scan_apply<<<n1 / 256, 256, 0, stream>>>(aArr, bArr, cry, hp);
    }

    // 4) output GEMM (WMMA) + residual + bias -> out
    {
        int tiles = (MM / (16 * MBLK)) * (DIN / 16);
        gemm_out<<<tiles / 4, 128, 0, stream>>>(hp, wop, x, ob, out);
    }
}
